// MHA_74964359185035
// MI455X (gfx1250) — matrix operand order repaired and checked by reading, not compiled, not run on hardware
//
#include <hip/hip_runtime.h>

// ---------------------------------------------------------------------------
// GPT-mini forward for MI455X (gfx1250): bf16 WMMA for all GEMMs + attention,
// async global->LDS tile staging (ASYNCcnt double buffering), f32 statistics.
// ---------------------------------------------------------------------------

#define Vv 13
#define Tt 128
#define Dd 512
#define Hh 8
#define Ee 64
#define Ll 6
#define Bb 512
#define Mrows (Bb * Tt)   // 65536

typedef __bf16 bf16_t;
typedef bf16_t v16bf __attribute__((ext_vector_type(16)));
typedef bf16_t v8bf  __attribute__((ext_vector_type(8)));
typedef float  v8f   __attribute__((ext_vector_type(8)));

__device__ __forceinline__ v8f vzero8() {
  v8f z;
#pragma unroll
  for (int i = 0; i < 8; ++i) z[i] = 0.0f;
  return z;
}

__device__ __forceinline__ v8f wmma_bf16(v16bf a, v16bf b, v8f c) {
  return __builtin_amdgcn_wmma_f32_16x16x32_bf16(
      false, a, false, b, (short)0, c, false, false);
}

__device__ __forceinline__ v16bf combine16(v8bf lo, v8bf hi) {
  return __builtin_shufflevector(lo, hi, 0, 1, 2, 3, 4, 5, 6, 7,
                                 8, 9, 10, 11, 12, 13, 14, 15);
}

// A fragment (16x32 bf16): m = lane&15; K = 16*(lane>=16) .. +15,
// one contiguous 32B run -> 2x ds_load_b128.
__device__ __forceinline__ v16bf load_a_frag(const bf16_t* tile, int ld) {
  const int lane = threadIdx.x & 31;
  const bf16_t* p = tile + (lane & 15) * ld + ((lane >> 4) << 4);
  return combine16(*(const v8bf*)p, *(const v8bf*)(p + 8));
}

// B fragment from a TRANSPOSED tile (row = output column n, ld elems of K):
// lane n holds K = 16*(lane>=16) .. +15, one contiguous 32B run.
__device__ __forceinline__ v16bf load_bt_frag(const bf16_t* tile, int ld) {
  const int lane = threadIdx.x & 31;
  const bf16_t* p = tile + (lane & 15) * ld + ((lane >> 4) << 4);
  return combine16(*(const v8bf*)p, *(const v8bf*)(p + 8));
}

// ---- CDNA5 async global->LDS (ASYNCcnt) ------------------------------------
__device__ __forceinline__ void async_b128(void* lds, const bf16_t* g) {
  unsigned off = (unsigned)(uintptr_t)lds;  // low 32 bits = LDS byte offset
  asm volatile("global_load_async_to_lds_b128 %0, %1, off"
               :: "v"(off), "v"(g) : "memory");
}
__device__ __forceinline__ void wait_async0() {
  asm volatile("s_wait_asynccnt 0" ::: "memory");
}
__device__ __forceinline__ void wait_async2() {
  asm volatile("s_wait_asynccnt 2" ::: "memory");
}
__device__ __forceinline__ void wait_async4() {
  asm volatile("s_wait_asynccnt 4" ::: "memory");
}

// ---------------------------------------------------------------------------
// Weight prep: repack to bf16 TRANSPOSED [N][K] row-major
// ---------------------------------------------------------------------------
__global__ void prep_wqkv_kernel(const float* __restrict__ W, bf16_t* __restrict__ out) {
  // out[(l*512 + c)*512 + d] = W[((l*H + c>>6)*D + d)*E + (c&63)]
  int idx = blockIdx.x * blockDim.x + threadIdx.x;  // over L*512*512
  int d = idx & 511;
  int c = (idx >> 9) & 511;
  int l = idx >> 18;
  out[idx] = (bf16_t)W[(((l * Hh + (c >> 6)) * Dd + d) << 6) + (c & 63)];
}

__global__ void prep_woutT_kernel(const float* __restrict__ W, bf16_t* __restrict__ out) {
  // out[(l*512 + n)*512 + d] = W[(l*512 + d)*512 + n]
  int idx = blockIdx.x * blockDim.x + threadIdx.x;
  int d = idx & 511;
  int n = (idx >> 9) & 511;
  int l = idx >> 18;
  out[idx] = (bf16_t)W[((size_t)l * Dd + d) * Dd + n];
}

__global__ void prep_wheadT_kernel(const float* __restrict__ W, bf16_t* __restrict__ out) {
  // out[n*512 + d] = Whead[d*13 + n] (n<13), 0 otherwise; 16x512
  int idx = blockIdx.x * blockDim.x + threadIdx.x;  // over 16*512
  int d = idx & 511;
  int n = idx >> 9;
  out[idx] = (n < Vv) ? (bf16_t)W[d * Vv + n] : (bf16_t)0.0f;
}

// ---------------------------------------------------------------------------
// Embedding
// ---------------------------------------------------------------------------
__global__ __launch_bounds__(256) void embed_kernel(
    const int* __restrict__ tok, const float* __restrict__ wte,
    const float* __restrict__ wpe, float* __restrict__ xf, bf16_t* __restrict__ xb) {
  const int row = blockIdx.x;
  const int t = row & (Tt - 1);
  const int tk = tok[row];
  const size_t base = (size_t)row * Dd;
#pragma unroll
  for (int i = 0; i < 2; ++i) {
    const int c = threadIdx.x + i * 256;
    const float v = wte[tk * Dd + c] + wpe[t * Dd + c];
    xf[base + c] = v;
    xb[base + c] = (bf16_t)v;
  }
}

// ---------------------------------------------------------------------------
// WMMA GEMM: C[M,512] = A[M,512](bf16) * BT^T + bias, BT is [512][512] (N-major)
// 256 threads, tile 128x128, K-step 32, async double-buffered staging.
// ---------------------------------------------------------------------------
__global__ __launch_bounds__(256) void gemm_bf16_kernel(
    const bf16_t* __restrict__ A, const bf16_t* __restrict__ BT,
    const float* __restrict__ bias, float* __restrict__ Cf, bf16_t* __restrict__ Cb) {
  __shared__ __align__(16) bf16_t As[2][128 * 40];
  __shared__ __align__(16) bf16_t Bs[2][128 * 40];
  const int tid = threadIdx.x;
  const int w = tid >> 5, lane = tid & 31, hi = lane >> 4, ln = lane & 15;
  const int m0 = blockIdx.y * 128, n0 = blockIdx.x * 128;
  const int r = tid >> 1, p = tid & 1;  // 2 threads/row, 16 elems (2x b128) each

  const bf16_t* gA = A + (size_t)(m0 + r) * Dd + p * 16;
  const bf16_t* gB = BT + (size_t)(n0 + r) * Dd + p * 16;

  v8f acc[8];
#pragma unroll
  for (int nt = 0; nt < 8; ++nt) acc[nt] = vzero8();

  // issue tile kt into buffer buf: 4 async b128 per thread (2 A + 2 B)
  auto issue = [&](int buf, int k0) {
    bf16_t* la = &As[buf][r * 40 + p * 16];
    bf16_t* lb = &Bs[buf][r * 40 + p * 16];
    async_b128(la, gA + k0);
    async_b128(la + 8, gA + k0 + 8);
    async_b128(lb, gB + k0);
    async_b128(lb + 8, gB + k0 + 8);
  };

  issue(0, 0);
#pragma unroll
  for (int kt = 0; kt < Dd / 32; ++kt) {
    const int buf = kt & 1;
    if (kt + 1 < Dd / 32) {
      issue(buf ^ 1, (kt + 1) * 32);
      wait_async4();  // loads retire in order: current tile's 4 are done
    } else {
      wait_async0();
    }
    __syncthreads();
    v16bf af = load_a_frag(&As[buf][w * 16 * 40], 40);
#pragma unroll
    for (int nt = 0; nt < 8; ++nt) {
      v16bf bfr = load_bt_frag(&Bs[buf][nt * 16 * 40], 40);
      acc[nt] = wmma_bf16(af, bfr, acc[nt]);
    }
    __syncthreads();
  }

#pragma unroll
  for (int nt = 0; nt < 8; ++nt) {
#pragma unroll
    for (int rr = 0; rr < 8; ++rr) {
      const int m = m0 + w * 16 + rr + hi * 8;
      const int n = n0 + nt * 16 + ln;
      float v = acc[nt][rr] + (bias ? bias[n] : 0.0f);
      if (Cf) Cf[(size_t)m * Dd + n] = v;
      if (Cb) Cb[(size_t)m * Dd + n] = (bf16_t)v;
    }
  }
}

// ---------------------------------------------------------------------------
// Fused attention per (b,h)
// ---------------------------------------------------------------------------
__global__ __launch_bounds__(256) void attn_kernel(
    const bf16_t* __restrict__ qb, const bf16_t* __restrict__ kb,
    const bf16_t* __restrict__ vb, bf16_t* __restrict__ ob) {
  __shared__ __align__(16) bf16_t smem[26112];
  bf16_t* qs = smem;                 // [128][72]
  bf16_t* ks = smem + 128 * 72;      // [128][72]
  bf16_t* atts = smem;               // [128][136] (reuses qs/ks after barrier)
  bf16_t* vsT = smem + 128 * 136;    // [64][136] transposed v

  const int blk = blockIdx.x;
  const int b = blk >> 3, h = blk & 7;
  const size_t row0 = (size_t)b * Tt;
  const int tid = threadIdx.x;
  const int w = tid >> 5, lane = tid & 31, hi = lane >> 4, ln = lane & 15;
  const int r = tid >> 1, p = tid & 1;

  {  // async-stage q,k tiles (128 rows x 64 bf16), 32 elems (4x b128)/thread each
    const bf16_t* gq = qb + (row0 + r) * Dd + h * Ee + p * 32;
    const bf16_t* gk = kb + (row0 + r) * Dd + h * Ee + p * 32;
    bf16_t* lq = qs + r * 72 + p * 32;
    bf16_t* lk = ks + r * 72 + p * 32;
#pragma unroll
    for (int i = 0; i < 4; ++i) {
      async_b128(lq + i * 8, gq + i * 8);
      async_b128(lk + i * 8, gk + i * 8);
    }
  }
  wait_async0();
  __syncthreads();

  // att = q * k^T : M=128, N=128, K=64; B column s = row s of k (contiguous)
  v8f acc[8];
#pragma unroll
  for (int nt = 0; nt < 8; ++nt) acc[nt] = vzero8();
#pragma unroll
  for (int kk = 0; kk < 64; kk += 32) {
    v16bf af = load_a_frag(qs + w * 16 * 72 + kk, 72);
#pragma unroll
    for (int nt = 0; nt < 8; ++nt) {
      const bf16_t* pb = ks + (nt * 16 + ln) * 72 + kk + 16 * hi;
      v16bf bfr = combine16(*(const v8bf*)pb, *(const v8bf*)(pb + 8));
      acc[nt] = wmma_bf16(af, bfr, acc[nt]);
    }
  }
  __syncthreads();  // qs/ks dead; smem reused below

  {  // stage v transposed: coalesced 128-bit global loads, b16 LDS scatter
    v8bf vv[4];
    const v8bf* gv = (const v8bf*)(vb + (row0 + r) * Dd + h * Ee + p * 32);
#pragma unroll
    for (int i = 0; i < 4; ++i) vv[i] = gv[i];
#pragma unroll
    for (int i = 0; i < 32; ++i) {
      const int e = p * 32 + i;
      vsT[e * 136 + r] = vv[i >> 3][i & 7];
    }
  }

  // faithful quirk: zero (not -inf) masked slots BEFORE scaling
  const float scale = 0.044194173824159216f;  // 1/sqrt(D=512)
#pragma unroll
  for (int rr = 0; rr < 8; ++rr) {
    const int mrow = w * 16 + rr + hi * 8;
    float vals[8];
#pragma unroll
    for (int nt = 0; nt < 8; ++nt) {
      const int col = nt * 16 + ln;
      float x = acc[nt][rr];
      vals[nt] = ((col <= mrow) ? x : 0.0f) * scale;
    }
    float mx = vals[0];
#pragma unroll
    for (int nt = 1; nt < 8; ++nt) mx = fmaxf(mx, vals[nt]);
#pragma unroll
    for (int o = 8; o > 0; o >>= 1) mx = fmaxf(mx, __shfl_xor(mx, o, 32));
    float sum = 0.0f;
#pragma unroll
    for (int nt = 0; nt < 8; ++nt) { float e = __expf(vals[nt] - mx); vals[nt] = e; sum += e; }
#pragma unroll
    for (int o = 8; o > 0; o >>= 1) sum += __shfl_xor(sum, o, 32);
    const float inv = 1.0f / sum;
#pragma unroll
    for (int nt = 0; nt < 8; ++nt)
      atts[mrow * 136 + nt * 16 + ln] = (bf16_t)(vals[nt] * inv);
  }
  __syncthreads();

  // o = att * v : M=128, N=64, K=128 (B fragments = rows of vsT, contiguous)
  v8f oacc[4];
#pragma unroll
  for (int nt = 0; nt < 4; ++nt) oacc[nt] = vzero8();
#pragma unroll
  for (int kt = 0; kt < 4; ++kt) {
    v16bf af = load_a_frag(atts + w * 16 * 136 + kt * 32, 136);
#pragma unroll
    for (int nt = 0; nt < 4; ++nt) {
      const bf16_t* pb = vsT + (nt * 16 + ln) * 136 + kt * 32 + 16 * hi;
      v16bf bfr = combine16(*(const v8bf*)pb, *(const v8bf*)(pb + 8));
      oacc[nt] = wmma_bf16(af, bfr, oacc[nt]);
    }
  }

#pragma unroll
  for (int nt = 0; nt < 4; ++nt) {
#pragma unroll
    for (int rr = 0; rr < 8; ++rr) {
      const int m = w * 16 + rr + hi * 8;
      ob[(row0 + m) * Dd + h * Ee + nt * 16 + ln] = (bf16_t)oacc[nt][rr];
    }
  }
}

// ---------------------------------------------------------------------------
// Residual + LayerNorm
// ---------------------------------------------------------------------------
__global__ __launch_bounds__(256) void resid_ln_kernel(
    const float* __restrict__ yf, float* __restrict__ xf, bf16_t* __restrict__ xb,
    const float* __restrict__ g, const float* __restrict__ bta) {
  __shared__ float red[256];
  const int row = blockIdx.x;
  const int tid = threadIdx.x;
  const size_t base = (size_t)row * Dd;
  const float v0 = yf[base + tid] + xf[base + tid];
  const float v1 = yf[base + tid + 256] + xf[base + tid + 256];
  red[tid] = v0 + v1;
  __syncthreads();
  for (int o = 128; o > 0; o >>= 1) { if (tid < o) red[tid] += red[tid + o]; __syncthreads(); }
  const float mu = red[0] * (1.0f / Dd);
  __syncthreads();
  const float d0 = v0 - mu, d1 = v1 - mu;
  red[tid] = d0 * d0 + d1 * d1;
  __syncthreads();
  for (int o = 128; o > 0; o >>= 1) { if (tid < o) red[tid] += red[tid + o]; __syncthreads(); }
  const float rstd = rsqrtf(red[0] * (1.0f / Dd) + 1e-5f);
  const float o0 = d0 * rstd * g[tid] + bta[tid];
  const float o1 = d1 * rstd * g[tid + 256] + bta[tid + 256];
  xf[base + tid] = o0;         xb[base + tid] = (bf16_t)o0;
  xf[base + tid + 256] = o1;   xb[base + tid + 256] = (bf16_t)o1;
}

// ---------------------------------------------------------------------------
// Head GEMM: logits[M,13] = x[M,512] * WheadT^T + bhead, WheadT is [16][512]
// ---------------------------------------------------------------------------
__global__ __launch_bounds__(256) void head_gemm_kernel(
    const bf16_t* __restrict__ A, const bf16_t* __restrict__ WhT,
    const float* __restrict__ bhead, float* __restrict__ logits) {
  __shared__ __align__(16) bf16_t As[2][128 * 40];
  __shared__ __align__(16) bf16_t Ws[16 * 512];
  const int tid = threadIdx.x;
  const int w = tid >> 5, lane = tid & 31, hi = lane >> 4, ln = lane & 15;
  const int m0 = blockIdx.x * 128;
  const int r = tid >> 1, p = tid & 1;

  {  // async-stage the whole head matrix (16KB): 32 elems (4x b128)/thread
    const bf16_t* gw = WhT + tid * 32;
    bf16_t* lw = Ws + tid * 32;
#pragma unroll
    for (int i = 0; i < 4; ++i) async_b128(lw + i * 8, gw + i * 8);
  }

  const bf16_t* gA = A + (size_t)(m0 + r) * Dd + p * 16;
  auto issueA = [&](int buf, int k0) {
    bf16_t* la = &As[buf][r * 40 + p * 16];
    async_b128(la, gA + k0);
    async_b128(la + 8, gA + k0 + 8);
  };

  v8f acc = vzero8();
  issueA(0, 0);
#pragma unroll
  for (int kt = 0; kt < Dd / 32; ++kt) {
    const int buf = kt & 1;
    if (kt + 1 < Dd / 32) {
      issueA(buf ^ 1, (kt + 1) * 32);
      wait_async2();  // retires Ws (first iter) + current A tile
    } else {
      wait_async0();
    }
    __syncthreads();
    v16bf af = load_a_frag(&As[buf][w * 16 * 40], 40);
    const bf16_t* pb = Ws + ln * 512 + kt * 32 + 16 * hi;
    v16bf bfr = combine16(*(const v8bf*)pb, *(const v8bf*)(pb + 8));
    acc = wmma_bf16(af, bfr, acc);
    __syncthreads();
  }
#pragma unroll
  for (int rr = 0; rr < 8; ++rr) {
    const int m = m0 + w * 16 + rr + hi * 8;
    if (ln < Vv) logits[(size_t)m * Vv + ln] = acc[rr] + bhead[ln];
  }
}

// ---------------------------------------------------------------------------
// Loss
// ---------------------------------------------------------------------------
__global__ void loss_kernel(const float* __restrict__ logits,
                            const int* __restrict__ label, float* __restrict__ out) {
  __shared__ float red[512];
  const int t = threadIdx.x;
  const float* row = logits + ((size_t)t * Tt + (Tt - 1)) * Vv;
  float mx = row[0];
#pragma unroll
  for (int i = 1; i < Vv; ++i) mx = fmaxf(mx, row[i]);
  float s = 0.0f;
#pragma unroll
  for (int i = 0; i < Vv; ++i) s += __expf(row[i] - mx);
  red[t] = (mx + __logf(s)) - row[label[t]];
  __syncthreads();
  for (int o = 256; o > 0; o >>= 1) { if (t < o) red[t] += red[t + o]; __syncthreads(); }
  if (t == 0) out[0] = red[0] * (1.0f / Bb);
}

// ---------------------------------------------------------------------------
extern "C" void kernel_launch(void* const* d_in, const int* in_sizes, int n_in,
                              void* d_out, int out_size, void* d_ws, size_t ws_size,
                              hipStream_t stream) {
  const int*   tok   = (const int*)d_in[0];
  const int*   label = (const int*)d_in[1];
  const float* wte   = (const float*)d_in[2];
  const float* wpe   = (const float*)d_in[3];
  const float* Wq    = (const float*)d_in[4];
  const float* bq    = (const float*)d_in[5];
  const float* Wk    = (const float*)d_in[6];
  const float* bk    = (const float*)d_in[7];
  const float* Wv    = (const float*)d_in[8];
  const float* bv    = (const float*)d_in[9];
  const float* Wout  = (const float*)d_in[10];
  const float* bout  = (const float*)d_in[11];
  const float* lng   = (const float*)d_in[12];
  const float* lnb   = (const float*)d_in[13];
  const float* Whead = (const float*)d_in[14];
  const float* bhead = (const float*)d_in[15];
  float* out = (float*)d_out;

  char* ws = (char*)d_ws;
  size_t off = 0;
  auto alloc = [&](size_t n) { size_t o = off; off = (off + n + 255) & ~(size_t)255; return o; };
  float*  xf  = (float*)(ws + alloc((size_t)Mrows * Dd * 4));
  bf16_t* xb  = (bf16_t*)(ws + alloc((size_t)Mrows * Dd * 2));
  bf16_t* qb_ = (bf16_t*)(ws + alloc((size_t)Mrows * Dd * 2));
  bf16_t* kb_ = (bf16_t*)(ws + alloc((size_t)Mrows * Dd * 2));
  bf16_t* vb_ = (bf16_t*)(ws + alloc((size_t)Mrows * Dd * 2));
  bf16_t* ob_ = (bf16_t*)(ws + alloc((size_t)Mrows * Dd * 2));
  float*  yf  = (float*)(ws + alloc((size_t)Mrows * Dd * 4));
  bf16_t* wqT = (bf16_t*)(ws + alloc((size_t)Ll * Dd * 512 * 2));
  bf16_t* wkT = (bf16_t*)(ws + alloc((size_t)Ll * Dd * 512 * 2));
  bf16_t* wvT = (bf16_t*)(ws + alloc((size_t)Ll * Dd * 512 * 2));
  bf16_t* woT = (bf16_t*)(ws + alloc((size_t)Ll * Dd * 512 * 2));
  bf16_t* whT = (bf16_t*)(ws + alloc((size_t)16 * Dd * 2));

  prep_wqkv_kernel<<<Ll * Dd * 512 / 256, 256, 0, stream>>>(Wq, wqT);
  prep_wqkv_kernel<<<Ll * Dd * 512 / 256, 256, 0, stream>>>(Wk, wkT);
  prep_wqkv_kernel<<<Ll * Dd * 512 / 256, 256, 0, stream>>>(Wv, wvT);
  prep_woutT_kernel<<<Ll * Dd * Dd / 256, 256, 0, stream>>>(Wout, woT);
  prep_wheadT_kernel<<<16 * Dd / 256, 256, 0, stream>>>(Whead, whT);

  embed_kernel<<<Mrows, 256, 0, stream>>>(tok, wte, wpe, xf, xb);

  const dim3 ggrid(Dd / 128, Mrows / 128);  // (4, 512)
  for (int l = 0; l < Ll; ++l) {
    const size_t wo = (size_t)l * Dd * 512;
    gemm_bf16_kernel<<<ggrid, 256, 0, stream>>>(xb, wqT + wo, bq + l * 512, nullptr, qb_);
    gemm_bf16_kernel<<<ggrid, 256, 0, stream>>>(xb, wkT + wo, bk + l * 512, nullptr, kb_);
    gemm_bf16_kernel<<<ggrid, 256, 0, stream>>>(xb, wvT + wo, bv + l * 512, nullptr, vb_);
    attn_kernel<<<Bb * Hh, 256, 0, stream>>>(qb_, kb_, vb_, ob_);
    gemm_bf16_kernel<<<ggrid, 256, 0, stream>>>(ob_, woT + wo, bout + l * 512, yf, nullptr);
    resid_ln_kernel<<<Mrows, 256, 0, stream>>>(yf, xf, xb, lng + l * 512, lnb + l * 512);
  }

  head_gemm_kernel<<<Mrows / 128, 256, 0, stream>>>(xb, whT, bhead, out);
  loss_kernel<<<1, 512, 0, stream>>>(out, label, out + (size_t)Bb * Tt * Vv);
}
